// QuantumAttentionEngine_20976620274067
// MI455X (gfx1250) — matrix operand order repaired and checked by reading, not compiled, not run on hardware
//
#include <hip/hip_runtime.h>
#include <math.h>

#define BATCH   2
#define NCELLS  2048
#define HCN     2048
#define HCDIM   11
#define DIM     1024
#define INDIM   512
#define NLAYER  2
#define NHEAD   8
#define HD      128
#define NBINJ   6
#define NBFR    10
#define NFRS    128
#define NMORPH  48

typedef __attribute__((ext_vector_type(16))) _Float16 v16h;
typedef __attribute__((ext_vector_type(8)))  _Float16 v8h;
typedef __attribute__((ext_vector_type(8)))  float    v8f;
typedef __attribute__((ext_vector_type(4)))  unsigned int uint4v;
typedef __attribute__((ext_vector_type(4)))  int int4v;

__device__ __forceinline__ v8f wmma16(v16h a, v16h b, v8f c) {
  // D = A(16x32 f16) x B(32x16 f16) + C(16x16 f32)
  return __builtin_amdgcn_wmma_f32_16x16x32_f16(false, a, false, b, (short)0, c, false, false);
}

// --- CDNA5 async global->LDS copy (ASYNCcnt-tracked), with safe fallback ----
__device__ __forceinline__ void async_cp16(const _Float16* g, _Float16* l) {
#if __has_builtin(__builtin_amdgcn_global_load_async_to_lds_b128)
  typedef __attribute__((address_space(1))) int4v gvec_t;   // global v4i
  typedef __attribute__((address_space(3))) int4v lvec_t;   // LDS v4i
  __builtin_amdgcn_global_load_async_to_lds_b128((gvec_t*)g, (lvec_t*)l, 0, 0);
#else
  *(uint4v*)l = *(const uint4v*)g;
#endif
}
__device__ __forceinline__ void async_wait0() {
#if __has_builtin(__builtin_amdgcn_s_wait_asynccnt)
  __builtin_amdgcn_s_wait_asynccnt(0);
#else
  asm volatile("s_wait_asynccnt 0x0" ::: "memory");
#endif
}

// ---------------------------------------------------------------------------
// Phase A: update_cells kernels (f32, bandwidth-bound)
// ---------------------------------------------------------------------------

__global__ void k_zero1(float* p, int n) {
  int i = threadIdx.x;
  if (i < n) p[i] = 0.f;
}

__global__ void k_xmean(const float* __restrict__ x, float* __restrict__ xm) {
  __shared__ float red[256];
  int t = threadIdx.x;
  float s = 0.f;
  for (int i = t; i < BATCH * INDIM; i += 256) s += x[i];
  red[t] = s; __syncthreads();
  for (int st = 128; st > 0; st >>= 1) { if (t < st) red[t] += red[t + st]; __syncthreads(); }
  if (t == 0) xm[0] = red[0] / (float)(BATCH * INDIM);
}

__global__ __launch_bounds__(256)
void k_walk(const float* __restrict__ inR, const float* __restrict__ inI,
            float* __restrict__ outR, float* __restrict__ outI,
            const float* __restrict__ coinR, const float* __restrict__ coinI,
            const float* __restrict__ xm, float* __restrict__ normAcc) {
  int idx = blockIdx.x * 256 + threadIdx.x;   // HCN*DIM threads
  int n = idx / DIM, d = idx % DIM;
  float c00r = coinR[0], c01r = coinR[1], c10r = coinR[2], c11r = coinR[3];
  float c00i = coinI[0], c01i = coinI[1], c10i = coinI[2], c11i = coinI[3];

  float a0r = inR[(size_t)(n * 2 + 0) * DIM + d], a0i = inI[(size_t)(n * 2 + 0) * DIM + d];
  float a1r = inR[(size_t)(n * 2 + 1) * DIM + d], a1i = inI[(size_t)(n * 2 + 1) * DIM + d];
  float n0r = c00r * a0r - c00i * a0i + c01r * a1r - c01i * a1i;
  float n0i = c00r * a0i + c00i * a0r + c01r * a1i + c01i * a1r;

  float n1r = 0.f, n1i = 0.f;
  for (int bi = 0; bi < HCDIM; bi++) {
    int m = n ^ (1 << bi);
    float b0r = inR[(size_t)(m * 2 + 0) * DIM + d], b0i = inI[(size_t)(m * 2 + 0) * DIM + d];
    float b1r = inR[(size_t)(m * 2 + 1) * DIM + d], b1i = inI[(size_t)(m * 2 + 1) * DIM + d];
    n1r += c10r * b0r - c10i * b0i + c11r * b1r - c11i * b1i;
    n1i += c10r * b0i + c10i * b0r + c11r * b1i + c11i * b1r;
  }
  n1r *= (1.f / HCDIM); n1i *= (1.f / HCDIM);

  float th = 0.1f * xm[0];
  float ct = cosf(th), st = sinf(th);
  float o0r = n0r * ct - n0i * st, o0i = n0r * st + n0i * ct;
  float o1r = n1r * ct - n1i * st, o1i = n1r * st + n1i * ct;
  outR[(size_t)(n * 2 + 0) * DIM + d] = o0r; outI[(size_t)(n * 2 + 0) * DIM + d] = o0i;
  outR[(size_t)(n * 2 + 1) * DIM + d] = o1r; outI[(size_t)(n * 2 + 1) * DIM + d] = o1i;

  float ss = o0r * o0r + o0i * o0i + o1r * o1r + o1i * o1i;
  __shared__ float red[256];
  red[threadIdx.x] = ss; __syncthreads();
  for (int s = 128; s > 0; s >>= 1) { if (threadIdx.x < s) red[threadIdx.x] += red[threadIdx.x + s]; __syncthreads(); }
  if (threadIdx.x == 0) atomicAdd(normAcc, red[0]);
}

__global__ void k_norm_scale(float* R, float* I, const float* normAcc) {
  int idx = blockIdx.x * 256 + threadIdx.x;   // HCN*2*DIM
  float den = 1.f / (sqrtf(normAcc[0]) + 1e-8f);
  R[idx] *= den; I[idx] *= den;
}

__global__ void k_inject_stats(const float* __restrict__ R, const float* __restrict__ I,
                               float* __restrict__ probs, float* __restrict__ ph,
                               float* __restrict__ ptot) {
  int n = blockIdx.x, t = threadIdx.x;
  float ps = 0.f, as = 0.f;
  for (int d = t; d < DIM; d += 256) {
    float r0 = R[(size_t)(n * 2 + 0) * DIM + d], i0 = I[(size_t)(n * 2 + 0) * DIM + d];
    float r1 = R[(size_t)(n * 2 + 1) * DIM + d], i1 = I[(size_t)(n * 2 + 1) * DIM + d];
    ps += r0 * r0 + i0 * i0 + r1 * r1 + i1 * i1;
    as += atan2f(i0 + i1, r0 + r1);
  }
  __shared__ float s1[256], s2[256];
  s1[t] = ps; s2[t] = as; __syncthreads();
  for (int st = 128; st > 0; st >>= 1) {
    if (t < st) { s1[t] += s1[t + st]; s2[t] += s2[t + st]; }
    __syncthreads();
  }
  if (t == 0) { probs[n] = s1[0]; ph[n] = s2[0] / (float)DIM; atomicAdd(ptot, s1[0]); }
}

__global__ void k_inject_apply(float* __restrict__ cells, const float* __restrict__ probs,
                               const float* __restrict__ ph, const float* __restrict__ ptot) {
  int idx = blockIdx.x * 256 + threadIdx.x;   // NCELLS * DIM/2
  int i = idx / (DIM / 2), d = idx % (DIM / 2);
  float inv = 1.f / (ptot[0] + 1e-8f);
  float p = probs[i] * inv;
  float phc = ph[i] * 0.3f;
  float sc = 0.8f + 0.4f * p;
  float h1 = cells[(size_t)i * DIM + d] * sc;
  float h2 = cells[(size_t)i * DIM + d + DIM / 2] * sc;
  float c = cosf(phc), s = sinf(phc);
  float v1 = 0.5f * (h1 * c - h2 * s) + 0.5f * h1;
  float v2 = 0.5f * (h1 * s + h2 * c) + 0.5f * h2;
  float interf = 0.f;
  for (int bi = 0; bi < NBINJ; bi++) {
    int j = i ^ (1 << bi);
    interf += (p - probs[j] * inv) * 0.03f;
  }
  cells[(size_t)i * DIM + d] = v1 + interf;
  cells[(size_t)i * DIM + d + DIM / 2] = v2 + interf;
}

__global__ __launch_bounds__(1024)
void k_frustration(float* __restrict__ cells, const float* __restrict__ fsig) {
  int d = threadIdx.x;
  for (int i = 0; i < NFRS; i++) {
    float fi = fsig[i];
    float infl = 0.f;
    for (int bi = 0; bi < NBFR; bi++) {
      int j = i ^ (1 << bi);
      infl += fi * fsig[j] * cells[(size_t)j * DIM + d];
    }
    cells[(size_t)i * DIM + d] = 0.85f * cells[(size_t)i * DIM + d] + 0.15f * infl * (1.f / NBFR);
  }
}

__global__ void k_standing(float* __restrict__ cells, const int* __restrict__ step) {
  int idx = blockIdx.x * 256 + threadIdx.x;
  int i = idx / DIM;
  float st = (float)step[0] * 0.15f;
  float fwd = fmodf(st, (float)NCELLS);
  float bwd = fmodf((float)NCELLS - st, (float)NCELLS);
  float a1 = 1.f / coshf(((float)i - fwd) * 0.5f);
  float a2 = 1.f / coshf(((float)i - bwd) * 0.5f);
  cells[idx] *= (1.f + 0.03f * (a1 * a1 + a2 * a2));
}

__global__ __launch_bounds__(1024)
void k_morphism(float* __restrict__ cells, const int* __restrict__ step) {
  if (step[0] % 3 != 0) return;
  int d = threadIdx.x;
  for (int i = 0; i < NMORPH; i++) {
    float ci = cells[(size_t)i * DIM + d];
    float ms = 0.f;
    for (int k = 0; k < NMORPH; k++) ms += tanhf(cells[(size_t)k * DIM + d] - ci);
    cells[(size_t)i * DIM + d] = 0.9f * ci + 0.1f * ms * (1.f / (NMORPH - 1));
  }
}

__global__ void k_fmean(const float* __restrict__ cells, float* __restrict__ fm) {
  int gid = blockIdx.x * 256 + threadIdx.x;  // 8*DIM
  int f = gid / DIM, d = gid % DIM;
  float s = 0.f;
  for (int r = 0; r < 256; r++) s += cells[(size_t)(f * 256 + r) * DIM + d];
  fm[gid] = s * (1.f / 256.f);
}

__global__ void k_fg(const float* __restrict__ fm, float* __restrict__ g) {
  int d = blockIdx.x * 256 + threadIdx.x;
  float s = 0.f;
  for (int f = 0; f < 8; f++) s += fm[f * DIM + d];
  g[d] = s * 0.125f;
}

__global__ void k_fapply(float* __restrict__ cells, const float* __restrict__ fm,
                         const float* __restrict__ g, const int* __restrict__ step) {
  int idx = blockIdx.x * 256 + threadIdx.x;
  int i = idx / DIM, d = idx % DIM;
  int f = i >> 8, r = i & 255;
  float v = 0.85f * cells[idx] + 0.15f * fm[f * DIM + d];
  if (step[0] > 5 && r < 64) v = 0.85f * v + 0.15f * g[d];
  cells[idx] = v;
}

// ---------------------------------------------------------------------------
// Phase B: transformer kernels (f16 operands, f32 accumulate, async tiles)
// ---------------------------------------------------------------------------

__global__ void k_cvt16(const float* __restrict__ s, _Float16* __restrict__ d, int n) {
  int i = blockIdx.x * 256 + threadIdx.x;
  if (i < n) d[i] = (_Float16)s[i];
}

__global__ void k_xproj(const float* __restrict__ x, const float* __restrict__ w_in,
                        const float* __restrict__ b_in, float* __restrict__ xp) {
  int gid = blockIdx.x * 256 + threadIdx.x;  // BATCH*DIM
  int b = gid / DIM, j = gid % DIM;
  float s = b_in[j];
  for (int k = 0; k < INDIM; k++) s += x[b * INDIM + k] * w_in[(size_t)j * INDIM + k];
  xp[gid] = s;
}

__global__ void k_act(const float* __restrict__ cells, const float* __restrict__ xp,
                      float* __restrict__ act, _Float16* __restrict__ acth) {
  int idx = blockIdx.x * 256 + threadIdx.x;  // BATCH*NCELLS*DIM
  int b = idx / (NCELLS * DIM);
  int nd = idx % (NCELLS * DIM);
  int d = nd % DIM;
  float v = cells[nd] + 0.1f * xp[b * DIM + d];
  act[idx] = v;
  acth[idx] = (_Float16)v;
}

// C(MxN) = A(MxK,f16) @ W(N,K,f16)^T + bias(f32); outputs f32 C and/or f16 Ch.
// 128 threads = 4 waves; 64x64 tile; double-buffered async LDS pipeline.
#define GM_BK 32
__device__ __forceinline__ void gemm_fill(_Float16 (*As)[GM_BK + 8], _Float16 (*Ws)[GM_BK + 8],
                                          const _Float16* __restrict__ A, int lda,
                                          const _Float16* __restrict__ W, int ldw,
                                          int mB, int nB, int k0, int tid) {
  // 64 rows x 32 halves = 4 x b128 chunks per row; 256 chunks per operand
  for (int e = tid; e < 64 * (GM_BK / 8); e += 128) {
    int r = e >> 2, c8 = (e & 3) * 8;
    async_cp16(&A[(size_t)(mB + r) * lda + k0 + c8], &As[r][c8]);
    async_cp16(&W[(size_t)(nB + r) * ldw + k0 + c8], &Ws[r][c8]);
  }
}

__global__ __launch_bounds__(128)
void k_gemm(const _Float16* __restrict__ A, int lda,
            const _Float16* __restrict__ W, int ldw,
            const float* __restrict__ bias,
            float* __restrict__ C, _Float16* __restrict__ Ch, int ldc,
            int M, int N, int K) {
  __shared__ _Float16 As[2][64][GM_BK + 8];
  __shared__ _Float16 Ws[2][64][GM_BK + 8];
  int nB = blockIdx.x * 64;
  int mB = blockIdx.y * 64;
  int tid = threadIdx.x;
  int wave = tid >> 5, lane = tid & 31;
  int lhalf = lane >> 4, l15 = lane & 15;
  v8f acc[4] = {};

  gemm_fill(As[0], Ws[0], A, lda, W, ldw, mB, nB, 0, tid);
  async_wait0();
  __syncthreads();

  int buf = 0;
  for (int k0 = 0; k0 < K; k0 += GM_BK) {
    int nxt = buf ^ 1;
    if (k0 + GM_BK < K)
      gemm_fill(As[nxt], Ws[nxt], A, lda, W, ldw, mB, nB, k0 + GM_BK, tid);

    v16h a;
    int arow = wave * 16 + l15;
    #pragma unroll
    for (int h = 0; h < 16; h++) {
      int kk = h + (lhalf << 4);
      a[h] = As[buf][arow][kk];
    }
    #pragma unroll
    for (int t = 0; t < 4; t++) {
      v16h b;
      #pragma unroll
      for (int h = 0; h < 16; h++) {
        int kk = h + (lhalf << 4);
        b[h] = Ws[buf][t * 16 + l15][kk];
      }
      acc[t] = wmma16(a, b, acc[t]);
    }
    async_wait0();
    __syncthreads();
    buf = nxt;
  }
  #pragma unroll
  for (int t = 0; t < 4; t++) {
    #pragma unroll
    for (int r = 0; r < 8; r++) {
      int m = mB + wave * 16 + (lhalf << 3) + r;
      int n = nB + t * 16 + l15;
      float v = acc[t][r] + bias[n];
      if (C)  C[(size_t)m * ldc + n] = v;
      if (Ch) Ch[(size_t)m * ldc + n] = (_Float16)v;
    }
  }
}

#define QSCALE 0.08838834764831845f  // 1/sqrt(HD); applied post-WMMA

// Build an A-fragment (16x32) from 16B-aligned global f16 row pointer.
__device__ __forceinline__ v16h load_afrag(const _Float16* p, int c, int lhalf) {
  const _Float16* p0 = p + c * 32 + (lhalf << 4);
  v8h lo = *(const v8h*)(p0);
  v8h hi = *(const v8h*)(p0 + 8);
  return __builtin_shufflevector(lo, hi, 0, 1, 2, 3, 4, 5, 6, 7,
                                 8, 9, 10, 11, 12, 13, 14, 15);
}

// Pass 1: per-(b,h,q) running max & exp-sum. stats[(b,h,q)] = {m, l}.
__global__ __launch_bounds__(128)
void k_attn_stats(const _Float16* __restrict__ qkvh, float* __restrict__ stats) {
  __shared__ _Float16 Ks[64][HD + 8];
  int qb = blockIdx.x & 31;
  int h  = (blockIdx.x >> 5) & 7;
  int b  = blockIdx.x >> 8;
  int tid = threadIdx.x, wave = tid >> 5, lane = tid & 31;
  int lhalf = lane >> 4, l15 = lane & 15;
  int qRow0 = qb * 64 + wave * 16;

  v16h aq[4];
  {
    const _Float16* Qp = qkvh + (size_t)(b * NCELLS + qRow0 + l15) * (3 * DIM) + h * HD;
    #pragma unroll
    for (int c = 0; c < 4; c++) aq[c] = load_afrag(Qp, c, lhalf);
  }
  float mi[8], li[8];
  #pragma unroll
  for (int r = 0; r < 8; r++) { mi[r] = -3.0e38f; li[r] = 0.f; }

  for (int kt = 0; kt < NCELLS; kt += 64) {
    for (int e = tid; e < 64 * (HD / 8); e += 128) {
      int r = e >> 4, c8 = (e & 15) * 8;
      async_cp16(&qkvh[(size_t)(b * NCELLS + kt + r) * (3 * DIM) + DIM + h * HD + c8], &Ks[r][c8]);
    }
    async_wait0();
    __syncthreads();
    v8f sc[4] = {};
    #pragma unroll
    for (int t = 0; t < 4; t++) {
      #pragma unroll
      for (int c = 0; c < 4; c++) {
        v16h bk;
        #pragma unroll
        for (int hh = 0; hh < 16; hh++) {
          int kk = c * 32 + hh + (lhalf << 4);
          bk[hh] = Ks[t * 16 + l15][kk];
        }
        sc[t] = wmma16(aq[c], bk, sc[t]);
      }
    }
    #pragma unroll
    for (int r = 0; r < 8; r++) {
      float s0 = sc[0][r] * QSCALE, s1 = sc[1][r] * QSCALE;
      float s2 = sc[2][r] * QSCALE, s3 = sc[3][r] * QSCALE;
      float mx = fmaxf(fmaxf(s0, s1), fmaxf(s2, s3));
      mx = fmaxf(mx, __shfl_xor(mx, 1, 32));
      mx = fmaxf(mx, __shfl_xor(mx, 2, 32));
      mx = fmaxf(mx, __shfl_xor(mx, 4, 32));
      mx = fmaxf(mx, __shfl_xor(mx, 8, 32));
      float nm = fmaxf(mi[r], mx);
      float es = __expf(s0 - nm) + __expf(s1 - nm) + __expf(s2 - nm) + __expf(s3 - nm);
      es += __shfl_xor(es, 1, 32);
      es += __shfl_xor(es, 2, 32);
      es += __shfl_xor(es, 4, 32);
      es += __shfl_xor(es, 8, 32);
      li[r] = li[r] * __expf(mi[r] - nm) + es;
      mi[r] = nm;
    }
    __syncthreads();
  }
  if (l15 == 0) {
    #pragma unroll
    for (int r = 0; r < 8; r++) {
      int q = qRow0 + (lhalf << 3) + r;
      size_t si = ((size_t)(b * NHEAD + h) * NCELLS + q) * 2;
      stats[si] = mi[r]; stats[si + 1] = li[r];
    }
  }
}

// Pass 2: O = softmax(QK^T) @ V with final stats; writes f16 (b,n,h*HD+d)
__global__ __launch_bounds__(128)
void k_attn_out(const _Float16* __restrict__ qkvh, const float* __restrict__ stats,
                _Float16* __restrict__ Oh) {
  __shared__ _Float16 Ks[64][HD + 8];
  __shared__ _Float16 Vs[64][HD + 8];
  __shared__ _Float16 Ps[4][16][64 + 8];
  int qb = blockIdx.x & 31;
  int h  = (blockIdx.x >> 5) & 7;
  int b  = blockIdx.x >> 8;
  int tid = threadIdx.x, wave = tid >> 5, lane = tid & 31;
  int lhalf = lane >> 4, l15 = lane & 15;
  int qRow0 = qb * 64 + wave * 16;

  v16h aq[4];
  {
    const _Float16* Qp = qkvh + (size_t)(b * NCELLS + qRow0 + l15) * (3 * DIM) + h * HD;
    #pragma unroll
    for (int c = 0; c < 4; c++) aq[c] = load_afrag(Qp, c, lhalf);
  }
  float mi[8], li[8];
  #pragma unroll
  for (int r = 0; r < 8; r++) {
    int q = qRow0 + (lhalf << 3) + r;
    size_t si = ((size_t)(b * NHEAD + h) * NCELLS + q) * 2;
    mi[r] = stats[si]; li[r] = stats[si + 1];
  }
  v8f ob[8] = {};

  for (int kt = 0; kt < NCELLS; kt += 64) {
    for (int e = tid; e < 64 * (HD / 8); e += 128) {
      int r = e >> 4, c8 = (e & 15) * 8;
      size_t base = (size_t)(b * NCELLS + kt + r) * (3 * DIM);
      async_cp16(&qkvh[base + DIM + h * HD + c8], &Ks[r][c8]);
      async_cp16(&qkvh[base + 2 * DIM + h * HD + c8], &Vs[r][c8]);
    }
    async_wait0();
    __syncthreads();
    v8f sc[4] = {};
    #pragma unroll
    for (int t = 0; t < 4; t++) {
      #pragma unroll
      for (int c = 0; c < 4; c++) {
        v16h bk;
        #pragma unroll
        for (int hh = 0; hh < 16; hh++) {
          int kk = c * 32 + hh + (lhalf << 4);
          bk[hh] = Ks[t * 16 + l15][kk];
        }
        sc[t] = wmma16(aq[c], bk, sc[t]);
      }
    }
    #pragma unroll
    for (int t = 0; t < 4; t++)
      #pragma unroll
      for (int r = 0; r < 8; r++) {
        float p = __expf(sc[t][r] * QSCALE - mi[r]) / li[r];
        Ps[wave][(lhalf << 3) + r][t * 16 + l15] = (_Float16)p;
      }
    v16h ap[2];
    #pragma unroll
    for (int c = 0; c < 2; c++)
      #pragma unroll
      for (int hh = 0; hh < 16; hh++) {
        int kk = c * 32 + hh + (lhalf << 4);
        ap[c][hh] = Ps[wave][l15][kk];
      }
    #pragma unroll
    for (int nt = 0; nt < 8; nt++) {
      #pragma unroll
      for (int c = 0; c < 2; c++) {
        v16h bv;
        #pragma unroll
        for (int hh = 0; hh < 16; hh++) {
          int kk = c * 32 + hh + (lhalf << 4);
          bv[hh] = Vs[kk][nt * 16 + l15];
        }
        ob[nt] = wmma16(ap[c], bv, ob[nt]);
      }
    }
    __syncthreads();
  }
  #pragma unroll
  for (int nt = 0; nt < 8; nt++)
    #pragma unroll
    for (int r = 0; r < 8; r++) {
      int q = qRow0 + (lhalf << 3) + r;
      int dd = nt * 16 + l15;
      Oh[(size_t)(b * NCELLS + q) * DIM + h * HD + dd] = (_Float16)ob[nt][r];
    }
}

// Tension second moment: sum over (b,q,k) of (mean_h p)^2.
__global__ __launch_bounds__(128)
void k_tension(const _Float16* __restrict__ qkvh, const float* __restrict__ stats,
               float* __restrict__ sum2_slot) {
  __shared__ _Float16 Qs[64][HD + 8];
  __shared__ _Float16 Ks[64][HD + 8];
  __shared__ float red[128];
  int qb = blockIdx.x & 31;
  int b  = blockIdx.x >> 5;
  int tid = threadIdx.x, wave = tid >> 5, lane = tid & 31;
  int lhalf = lane >> 4, l15 = lane & 15;
  int qRow0 = qb * 64 + wave * 16;
  float s2 = 0.f;

  for (int kt = 0; kt < NCELLS; kt += 64) {
    v8f macc[4] = {};
    for (int h = 0; h < NHEAD; h++) {
      __syncthreads();
      for (int e = tid; e < 64 * (HD / 8); e += 128) {
        int r = e >> 4, c8 = (e & 15) * 8;
        async_cp16(&qkvh[(size_t)(b * NCELLS + qb * 64 + r) * (3 * DIM) + h * HD + c8], &Qs[r][c8]);
        async_cp16(&qkvh[(size_t)(b * NCELLS + kt + r) * (3 * DIM) + DIM + h * HD + c8], &Ks[r][c8]);
      }
      async_wait0();
      __syncthreads();
      v16h aq[4];
      #pragma unroll
      for (int c = 0; c < 4; c++)
        #pragma unroll
        for (int hh = 0; hh < 16; hh++) {
          int kk = c * 32 + hh + (lhalf << 4);
          aq[c][hh] = Qs[wave * 16 + l15][kk];
        }
      float mi[8], li[8];
      #pragma unroll
      for (int r = 0; r < 8; r++) {
        int q = qRow0 + (lhalf << 3) + r;
        size_t si = ((size_t)(b * NHEAD + h) * NCELLS + q) * 2;
        mi[r] = stats[si]; li[r] = stats[si + 1];
      }
      v8f sc[4] = {};
      #pragma unroll
      for (int t = 0; t < 4; t++) {
        #pragma unroll
        for (int c = 0; c < 4; c++) {
          v16h bk;
          #pragma unroll
          for (int hh = 0; hh < 16; hh++) {
            int kk = c * 32 + hh + (lhalf << 4);
            bk[hh] = Ks[t * 16 + l15][kk];
          }
          sc[t] = wmma16(aq[c], bk, sc[t]);
        }
      }
      #pragma unroll
      for (int t = 0; t < 4; t++)
        #pragma unroll
        for (int r = 0; r < 8; r++)
          macc[t][r] += __expf(sc[t][r] * QSCALE - mi[r]) / li[r];
    }
    #pragma unroll
    for (int t = 0; t < 4; t++)
      #pragma unroll
      for (int r = 0; r < 8; r++) {
        float m = macc[t][r] * (1.f / NHEAD);
        s2 += m * m;
      }
  }
  red[tid] = s2; __syncthreads();
  for (int st = 64; st > 0; st >>= 1) { if (tid < st) red[tid] += red[tid + st]; __syncthreads(); }
  if (tid == 0) atomicAdd(sum2_slot, red[0]);
}

__global__ __launch_bounds__(256)
void k_ln(float* __restrict__ act, const float* __restrict__ proj,
          const float* __restrict__ g, const float* __restrict__ bb,
          _Float16* __restrict__ acth) {
  __shared__ float red[256];
  __shared__ float mu_s, var_s;
  int row = blockIdx.x, t = threadIdx.x;
  float v[4];
  float s = 0.f;
  #pragma unroll
  for (int i = 0; i < 4; i++) {
    int d = t + i * 256;
    v[i] = act[(size_t)row * DIM + d] + proj[(size_t)row * DIM + d];
    s += v[i];
  }
  red[t] = s; __syncthreads();
  for (int st = 128; st > 0; st >>= 1) { if (t < st) red[t] += red[t + st]; __syncthreads(); }
  if (t == 0) mu_s = red[0] / (float)DIM;
  __syncthreads();
  float mu = mu_s, s2 = 0.f;
  #pragma unroll
  for (int i = 0; i < 4; i++) { float dd = v[i] - mu; s2 += dd * dd; }
  red[t] = s2; __syncthreads();
  for (int st = 128; st > 0; st >>= 1) { if (t < st) red[t] += red[t + st]; __syncthreads(); }
  if (t == 0) var_s = red[0] / (float)DIM;
  __syncthreads();
  float inv = rsqrtf(var_s + 1e-5f);
  #pragma unroll
  for (int i = 0; i < 4; i++) {
    int d = t + i * 256;
    float o = (v[i] - mu) * inv * g[d] + bb[d];
    act[(size_t)row * DIM + d] = o;
    acth[(size_t)row * DIM + d] = (_Float16)o;
  }
}

__global__ void k_finalize(const float* __restrict__ sum2, float* __restrict__ out_t) {
  double n = (double)BATCH * NCELLS * NCELLS;
  double mean = 1.0 / (double)NCELLS;   // rows of softmax sum to 1
  float t = 0.f;
  for (int l = 0; l < NLAYER; l++) {
    double var = ((double)sum2[l] - n * mean * mean) / (n - 1.0);
    if (var < 0.0) var = 0.0;
    t += (float)sqrt(var);
  }
  out_t[0] = t / (float)NLAYER;
}

// ---------------------------------------------------------------------------
// Host orchestration
// ---------------------------------------------------------------------------

extern "C" void kernel_launch(void* const* d_in, const int* in_sizes, int n_in,
                              void* d_out, int out_size, void* d_ws, size_t ws_size,
                              hipStream_t stream) {
  const float* x    = (const float*)d_in[0];
  const float* ar   = (const float*)d_in[1];
  const float* ai   = (const float*)d_in[2];
  const float* cnr  = (const float*)d_in[3];
  const float* cni  = (const float*)d_in[4];
  const float* cemb = (const float*)d_in[5];
  const float* fsig = (const float*)d_in[6];
  const float* w_in = (const float*)d_in[7];
  const float* b_in = (const float*)d_in[8];
  const float* aiw  = (const float*)d_in[9];
  const float* aib  = (const float*)d_in[10];
  const float* aow  = (const float*)d_in[11];
  const float* aob  = (const float*)d_in[12];
  const float* lng  = (const float*)d_in[13];
  const float* lnb  = (const float*)d_in[14];
  const float* wout = (const float*)d_in[15];
  const float* bout = (const float*)d_in[16];
  const int*   step = (const int*)d_in[17];
  float* out = (float*)d_out;

  char* base = (char*)d_ws;
  size_t off = 0;
  auto alloc = [&](size_t bytes) -> char* {
    off = (off + 255) & ~(size_t)255;
    char* p = base + off;
    off += bytes;
    return p;
  };
  const size_t AMPS = (size_t)HCN * 2 * DIM;
  float* A0R   = (float*)alloc(AMPS * 4);
  float* A0I   = (float*)alloc(AMPS * 4);
  float* A1R   = (float*)alloc(AMPS * 4);
  float* A1I   = (float*)alloc(AMPS * 4);
  float* cells = (float*)alloc((size_t)NCELLS * DIM * 4);
  float* probs = (float*)alloc(NCELLS * 4);
  float* ph    = (float*)alloc(NCELLS * 4);
  float* fm    = (float*)alloc(8 * DIM * 4);
  float* gm    = (float*)alloc(DIM * 4);
  float* xp    = (float*)alloc(BATCH * DIM * 4);
  float* scal  = (float*)alloc(64 * 4);  // 0:xmean 1:norm 2:ptot 3:sum2_l0 4:sum2_l1
  float* act   = (float*)alloc((size_t)BATCH * NCELLS * DIM * 4);
  float* proj  = (float*)alloc((size_t)BATCH * NCELLS * DIM * 4);
  float* stats = (float*)alloc((size_t)BATCH * NHEAD * NCELLS * 2 * 4);
  _Float16* acth   = (_Float16*)alloc((size_t)BATCH * NCELLS * DIM * 2);
  _Float16* qkvh   = (_Float16*)alloc((size_t)BATCH * NCELLS * 3 * DIM * 2);
  _Float16* attnOh = (_Float16*)alloc((size_t)BATCH * NCELLS * DIM * 2);
  _Float16* wh     = (_Float16*)alloc((size_t)3 * DIM * DIM * 2);

  k_zero1<<<1, 64, 0, stream>>>(scal, 8);
  k_xmean<<<1, 256, 0, stream>>>(x, &scal[0]);
  (void)hipMemcpyAsync(cells, cemb, (size_t)NCELLS * DIM * sizeof(float),
                       hipMemcpyDeviceToDevice, stream);

  // walk step 1 + inject
  k_walk<<<(HCN * DIM) / 256, 256, 0, stream>>>(ar, ai, A0R, A0I, cnr, cni, &scal[0], &scal[1]);
  k_norm_scale<<<(HCN * 2 * DIM) / 256, 256, 0, stream>>>(A0R, A0I, &scal[1]);
  k_inject_stats<<<HCN, 256, 0, stream>>>(A0R, A0I, probs, ph, &scal[2]);
  k_inject_apply<<<(NCELLS * (DIM / 2)) / 256, 256, 0, stream>>>(cells, probs, ph, &scal[2]);

  // walk step 2 + inject
  k_zero1<<<1, 64, 0, stream>>>(&scal[1], 2);
  k_walk<<<(HCN * DIM) / 256, 256, 0, stream>>>(A0R, A0I, A1R, A1I, cnr, cni, &scal[0], &scal[1]);
  k_norm_scale<<<(HCN * 2 * DIM) / 256, 256, 0, stream>>>(A1R, A1I, &scal[1]);
  k_inject_stats<<<HCN, 256, 0, stream>>>(A1R, A1I, probs, ph, &scal[2]);
  k_inject_apply<<<(NCELLS * (DIM / 2)) / 256, 256, 0, stream>>>(cells, probs, ph, &scal[2]);

  k_frustration<<<1, 1024, 0, stream>>>(cells, fsig);
  k_standing<<<(NCELLS * DIM) / 256, 256, 0, stream>>>(cells, step);
  k_morphism<<<1, 1024, 0, stream>>>(cells, step);
  k_fmean<<<(8 * DIM) / 256, 256, 0, stream>>>(cells, fm);
  k_fg<<<DIM / 256, 256, 0, stream>>>(fm, gm);
  k_fapply<<<(NCELLS * DIM) / 256, 256, 0, stream>>>(cells, fm, gm, step);

  // transformer
  k_xproj<<<(BATCH * DIM) / 256, 256, 0, stream>>>(x, w_in, b_in, xp);
  k_act<<<(BATCH * NCELLS * DIM) / 256, 256, 0, stream>>>(cells, xp, act, acth);

  const int M = BATCH * NCELLS;   // 4096
  for (int l = 0; l < NLAYER; l++) {
    k_cvt16<<<(3 * DIM * DIM + 255) / 256, 256, 0, stream>>>(
        aiw + (size_t)l * 3 * DIM * DIM, wh, 3 * DIM * DIM);
    k_gemm<<<dim3((3 * DIM) / 64, M / 64), 128, 0, stream>>>(
        acth, DIM, wh, DIM, aib + (size_t)l * 3 * DIM,
        nullptr, qkvh, 3 * DIM, M, 3 * DIM, DIM);
    k_attn_stats<<<BATCH * NHEAD * (NCELLS / 64), 128, 0, stream>>>(qkvh, stats);
    k_attn_out<<<BATCH * NHEAD * (NCELLS / 64), 128, 0, stream>>>(qkvh, stats, attnOh);
    k_tension<<<BATCH * (NCELLS / 64), 128, 0, stream>>>(qkvh, stats, &scal[3 + l]);
    k_cvt16<<<(DIM * DIM + 255) / 256, 256, 0, stream>>>(
        aow + (size_t)l * DIM * DIM, wh, DIM * DIM);
    k_gemm<<<dim3(DIM / 64, M / 64), 128, 0, stream>>>(
        attnOh, DIM, wh, DIM, aob + (size_t)l * DIM,
        proj, nullptr, DIM, M, DIM, DIM);
    k_ln<<<M, 256, 0, stream>>>(act, proj, lng + l * DIM, lnb + l * DIM, acth);
  }
  k_cvt16<<<(INDIM * DIM + 255) / 256, 256, 0, stream>>>(wout, wh, INDIM * DIM);
  k_gemm<<<dim3(INDIM / 64, M / 64), 128, 0, stream>>>(
      acth, DIM, wh, DIM, bout, out, nullptr, INDIM, M, INDIM, DIM);
  k_finalize<<<1, 1, 0, stream>>>(&scal[3], out + (size_t)BATCH * NCELLS * INDIM);
}
